// LSM_3298534883781
// MI455X (gfx1250) — matrix operand order repaired and checked by reading, not compiled, not run on hardware
//
#include <hip/hip_runtime.h>

// ---------------------------------------------------------------------------
// LSM / spiking reservoir on MI455X (gfx1250, wave32).
//
//   in_cur = x @ W_in^T                      (once, WMMA f16->f32)
//   25x:  mem = 0.95*mem + in_cur + spk@W_rec^T - spk ; spk = (mem > 1)
//   out = sum(spk)/25
//
// Recurrent kernel: 32 batch rows per WG for all 25 steps (256 WGs).
// spk (exact {0,1} in f16) lives in LDS in WMMA A-fragment layout; mem lives
// in the WMMA accumulators. W_rec^T is fed by the Tensor Data Mover:
// per-wave 32x64 f16 slabs DMA'd into LDS (TENSOR_LOAD_TO_LDS, TENSORcnt),
// triple-buffered, so the WMMA chain waits on s_wait_tensorcnt instead of
// per-load s_wait_loadcnt.
// ---------------------------------------------------------------------------

typedef __attribute__((ext_vector_type(16))) _Float16 v16h;
typedef __attribute__((ext_vector_type(8)))  _Float16 v8h;
typedef __attribute__((ext_vector_type(8)))  float    v8f;
typedef __attribute__((ext_vector_type(4)))  unsigned v4u;
typedef __attribute__((ext_vector_type(4)))  int      v4i;
typedef __attribute__((ext_vector_type(8)))  int      v8i;

#define BATCH   8192
#define NIN     784
#define KINP    800        // 784 padded to multiple of 32
#define NRES    1000
#define NPAD    1024       // 1000 padded (multiple of 32)
#define TSTEPS  25
#define BETA    0.95f
#define BM      32         // batch rows per workgroup (recurrent kernel)
#define SLABH   2048       // 32 rows x 64 cols f16 per wave slab

// ---------------- prep: f32 -> f16 convert / transpose / zero-pad ----------

__global__ void cvt_x_f16(const float* __restrict__ x, _Float16* __restrict__ x16) {
    int i = blockIdx.x * 256 + threadIdx.x;          // over 8192*800
    if (i >= BATCH * KINP) return;
    int r = i / KINP, c = i % KINP;
    x16[i] = (c < NIN) ? (_Float16)x[r * NIN + c] : (_Float16)0.0f;
}

__global__ void cvt_win_t(const float* __restrict__ W, _Float16* __restrict__ T) {
    int i = blockIdx.x * 256 + threadIdx.x;          // T[k][n] = W_in[n][k]
    if (i >= KINP * NPAD) return;
    int k = i / NPAD, n = i % NPAD;
    T[i] = (k < NIN && n < NRES) ? (_Float16)W[n * NIN + k] : (_Float16)0.0f;
}

__global__ void cvt_wrec_t(const float* __restrict__ W, _Float16* __restrict__ T) {
    int i = blockIdx.x * 256 + threadIdx.x;          // T[k][n] = W_rec[n][k]
    if (i >= NPAD * NPAD) return;
    int k = i >> 10, n = i & (NPAD - 1);
    T[i] = (k < NRES && n < NRES) ? (_Float16)W[n * NRES + k] : (_Float16)0.0f;
}

// ---------------- in_cur = x @ W_in^T  (WMMA f16 -> f32) -------------------

__global__ __launch_bounds__(256) void incur_gemm(
    const _Float16* __restrict__ X,      // [8192][800]
    const _Float16* __restrict__ WinT,   // [800][1024]
    float* __restrict__ incur)           // [8192][1024]
{
    const int tid  = threadIdx.x;
    const int wave = tid >> 5;
    const int lane = tid & 31;
    const int hf   = lane >> 4;
    const int lrow = lane & 15;
    const int m0   = blockIdx.x * 16;
    const int nbase = wave * 128;

    v8f acc[8];
    #pragma unroll
    for (int nt = 0; nt < 8; nt++)
        #pragma unroll
        for (int r = 0; r < 8; r++) acc[nt][r] = 0.0f;

    #pragma unroll 2
    for (int kb = 0; kb < KINP / 32; kb++) {
        const int k0 = kb * 32;
        const _Float16* ap = X + (m0 + lrow) * KINP + k0 + hf * 8;
        union { v16h v; v8h h[2]; } a;
        a.h[0] = *(const v8h*)(ap);
        a.h[1] = *(const v8h*)(ap + 16);
        #pragma unroll
        for (int nt = 0; nt < 8; nt++) {
            v16h b;
            #pragma unroll
            for (int s = 0; s < 16; s++)
                b[s] = WinT[(k0 + (s & 7) + 8 * hf + 16 * (s >> 3)) * NPAD + nbase + nt * 16 + lrow];
            acc[nt] = __builtin_amdgcn_wmma_f32_16x16x32_f16(
                          false, a.v, false, b, (short)0, acc[nt], false, false);
        }
    }

    #pragma unroll
    for (int nt = 0; nt < 8; nt++)
        #pragma unroll
        for (int r = 0; r < 8; r++)
            incur[(m0 + r + 8 * hf) * NPAD + nbase + nt * 16 + lrow] = acc[nt][r];
}

// ---------------- TDM: DMA one 32x64 f16 tile of W_rec^T into LDS ----------

__device__ __forceinline__ void tdm_load_slab(const _Float16* gsrc, unsigned lds_addr) {
    unsigned long long ga = (unsigned long long)gsrc;
    v4u g0;
    g0.x = 1u;                                           // count=1, user descriptor
    g0.y = lds_addr;                                     // LDS byte address
    g0.z = (unsigned)ga;                                 // global_addr[31:0]
    g0.w = ((unsigned)(ga >> 32) & 0x01FFFFFFu) | (2u << 30);  // [56:32] | type=2
    v8i g1 = { 0x00010000,   // wg_mask=0, data_size=1 (2B), no pad/iterate/barrier
               0x04000000,   // tensor_dim0 = 1024   (bits 63:48)
               0x04000000,   // tensor_dim1 = 1024   (bits 95:80)
               0x00400000,   // tile_dim0   = 64     (bits 127:112)
               0x00000020,   // tile_dim1   = 32     (bits 143:128)
               0x00000400,   // tensor_dim0_stride = 1024 (bits 191:160)
               0x04000000,   // tensor_dim1_stride = 1024 (bits 223:208)
               0 };
    v4i z4 = {0, 0, 0, 0};
#if __clang_major__ >= 23
    v8i z8 = {0, 0, 0, 0, 0, 0, 0, 0};
    __builtin_amdgcn_tensor_load_to_lds(g0, g1, z4, z4, z8, 0);
#else
    __builtin_amdgcn_tensor_load_to_lds(g0, g1, z4, z4, 0);
#endif
}

// ---------------- 25 recurrent steps, state resident in VGPRs + LDS --------
// 512 threads = 16 waves. Wave w owns n-slab [w*64, w*64+64), MT=2 x NT=4.
// B slabs stream through 3 wave-private LDS buffers via the TDM.

__global__ __launch_bounds__(512) void lsm_steps(
    const _Float16* __restrict__ Brec,   // [1024][1024] = W_rec^T, f16, padded
    const float*    __restrict__ incur,  // [8192][1024]
    float*          __restrict__ out)    // [8192][1000]
{
    __shared__ _Float16 spk[BM * NPAD];            // 64 KB, A-fragment layout
    __shared__ _Float16 bstage[16 * 3 * SLABH];    // 192 KB, B slab ring buffers

    const int tid   = threadIdx.x;
    const int wave  = tid >> 5;          // 0..15 -> n-slab
    const int lane  = tid & 31;
    const int hf    = lane >> 4;
    const int lrow  = lane & 15;
    const int row0  = blockIdx.x * BM;
    const int nbase = wave * 64;

    // zero initial spikes: 16384 dwords / 512 threads
    {
        unsigned* s32 = (unsigned*)spk;
        #pragma unroll
        for (int i = 0; i < 32; i++) s32[tid + i * 512] = 0u;
    }

    // wave-private slab ring
    const _Float16* b0 = &bstage[(wave * 3 + 0) * SLABH];
    const _Float16* b1 = &bstage[(wave * 3 + 1) * SLABH];
    const _Float16* b2 = &bstage[(wave * 3 + 2) * SLABH];
    const _Float16* gB = Brec + nbase;             // this wave's 64-col slice

    // prime the TDM pipeline: slabs 0 and 1 of step 0
    tdm_load_slab(gB + 0 * 32 * NPAD, (unsigned)(unsigned long long)b0);
    tdm_load_slab(gB + 1 * 32 * NPAD, (unsigned)(unsigned long long)b1);

    __syncthreads();

    v8f mem[2][4];                       // doubles as WMMA accumulator
    unsigned cnt[2][4][2];               // spike counts, 4 x u8 per dword
    #pragma unroll
    for (int mt = 0; mt < 2; mt++)
        #pragma unroll
        for (int nt = 0; nt < 4; nt++) {
            #pragma unroll
            for (int r = 0; r < 8; r++) mem[mt][nt][r] = 0.0f;
            cnt[mt][nt][0] = 0u; cnt[mt][nt][1] = 0u;
        }

    for (int t = 0; t < TSTEPS; t++) {
        // phase 1: acc init = beta*mem + in_cur - spk_old (reset == prev spike)
        #pragma unroll
        for (int mt = 0; mt < 2; mt++)
            #pragma unroll
            for (int nt = 0; nt < 4; nt++)
                #pragma unroll
                for (int r = 0; r < 8; r++) {
                    int row = row0 + mt * 16 + r + 8 * hf;
                    float ic = incur[row * NPAD + nbase + nt * 16 + lrow];
                    float m  = mem[mt][nt][r];
                    float s  = (m > 1.0f) ? 1.0f : 0.0f;
                    mem[mt][nt][r] = BETA * m - s + ic;
                }
        // phase 2: mem += spk @ W_rec^T, B slabs streamed by the TDM
        #pragma unroll 2
        for (int kb = 0; kb < 32; kb++) {
            // keep 2 slabs in flight; wrap into next step's slabs at the tail
            int nk = kb + 2; if (nk >= 32) nk -= 32;
            bool more = (kb < 30) || (t + 1 < TSTEPS);
            if (more) {
                tdm_load_slab(gB + nk * 32 * NPAD, (unsigned)(unsigned long long)b2);
                __builtin_amdgcn_s_wait_tensorcnt(2);   // current slab resident
            } else if (kb == 30) {
                __builtin_amdgcn_s_wait_tensorcnt(1);
            } else {
                __builtin_amdgcn_s_wait_tensorcnt(0);
            }
            const int k0 = kb * 32;
            union { v16h v; v8h h[2]; } a[2];
            #pragma unroll
            for (int mt = 0; mt < 2; mt++) {
                const _Float16* ap = &spk[(mt * 16 + lrow) * NPAD + k0 + hf * 8];
                a[mt].h[0] = *(const v8h*)(ap);          // ds_load_b128
                a[mt].h[1] = *(const v8h*)(ap + 16);
            }
            #pragma unroll
            for (int nt = 0; nt < 4; nt++) {
                v16h b;                                  // slab row = K
                #pragma unroll
                for (int s = 0; s < 16; s++)
                    b[s] = b0[((s & 7) + 8 * hf + 16 * (s >> 3)) * 64 + nt * 16 + lrow];
                #pragma unroll
                for (int mt = 0; mt < 2; mt++)
                    mem[mt][nt] = __builtin_amdgcn_wmma_f32_16x16x32_f16(
                        false, a[mt].v, false, b, (short)0, mem[mt][nt], false, false);
            }
            const _Float16* tmp = b0; b0 = b1; b1 = b2; b2 = tmp;   // rotate ring
        }
        __syncthreads();                 // all spike reads complete
        // phase 3: spike, count (packed u8), publish to LDS
        #pragma unroll
        for (int mt = 0; mt < 2; mt++)
            #pragma unroll
            for (int nt = 0; nt < 4; nt++)
                #pragma unroll
                for (int r = 0; r < 8; r++) {
                    bool s = mem[mt][nt][r] > 1.0f;
                    cnt[mt][nt][r >> 2] += s ? (1u << (8 * (r & 3))) : 0u;
                    spk[(mt * 16 + r + 8 * hf) * NPAD + nbase + nt * 16 + lrow] =
                        s ? (_Float16)1.0f : (_Float16)0.0f;
                }
        __syncthreads();                 // new spikes visible
    }

    const float inv = 1.0f / (float)TSTEPS;
    #pragma unroll
    for (int mt = 0; mt < 2; mt++)
        #pragma unroll
        for (int nt = 0; nt < 4; nt++) {
            const int col = nbase + nt * 16 + lrow;
            if (col < NRES) {
                #pragma unroll
                for (int r = 0; r < 8; r++) {
                    unsigned c = (cnt[mt][nt][r >> 2] >> (8 * (r & 3))) & 0xFFu;
                    out[(row0 + mt * 16 + r + 8 * hf) * NRES + col] = (float)c * inv;
                }
            }
        }
}

// ---------------------------------------------------------------------------

extern "C" void kernel_launch(void* const* d_in, const int* in_sizes, int n_in,
                              void* d_out, int out_size, void* d_ws, size_t ws_size,
                              hipStream_t stream) {
    const float* x     = (const float*)d_in[0];   // [8192,784]
    const float* W_in  = (const float*)d_in[1];   // [1000,784]
    const float* W_rec = (const float*)d_in[2];   // [1000,1000]
    float* out = (float*)d_out;                   // [8192,1000]

    char* ws = (char*)d_ws;
    _Float16* x16   = (_Float16*)(ws);                                   // 13107200 B
    _Float16* WinT  = (_Float16*)(ws + 13107200);                        //  1638400 B
    _Float16* BrecT = (_Float16*)(ws + 13107200 + 1638400);              //  2097152 B
    float*    incur = (float*)   (ws + 13107200 + 1638400 + 2097152);    // 33554432 B
    (void)ws_size; (void)in_sizes; (void)n_in; (void)out_size;

    cvt_x_f16  <<<(BATCH * KINP + 255) / 256, 256, 0, stream>>>(x, x16);
    cvt_win_t  <<<(KINP * NPAD + 255) / 256, 256, 0, stream>>>(W_in, WinT);
    cvt_wrec_t <<<(NPAD * NPAD + 255) / 256, 256, 0, stream>>>(W_rec, BrecT);
    incur_gemm <<<BATCH / 16, 256, 0, stream>>>(x16, WinT, incur);
    lsm_steps  <<<BATCH / BM, 512, 0, stream>>>(BrecT, incur, out);
}
